// ComplexAttention_15075335209501
// MI455X (gfx1250) — compile-verified, matrix operand order checked by reading, not run on hardware
//
#include <hip/hip_runtime.h>

// ---------------------------------------------------------------------------
// Complex attention for MI455X (gfx1250), bf16 WMMA path.
//  - one-time weight convert/transpose: W (KxN fp32) -> Wt (NxK bf16)
//  - double-buffered, software-pipelined WMMA GEMM (64x128 tile, 8 waves)
//  - flash attention with per-wave 16-row query tiles, WMMA for S and P*V
// ---------------------------------------------------------------------------

typedef __bf16 bf16;
typedef __attribute__((ext_vector_type(16))) __bf16 v16bf;
typedef __attribute__((ext_vector_type(8)))  __bf16 v8bf;
typedef __attribute__((ext_vector_type(8)))  float  v8f;

#define D_MODEL   1024
#define NUM_HEADS 16
#define HEAD_DIM  64
#define BATCH     2
#define SEQ       2048
#define BT        (BATCH * SEQ)   // 4096 rows

union V16U { v16bf v; v8bf h[2]; };

// One 16-element bf16 fragment: two contiguous 8xbf16 (16B) chunks at p and
// p+16 elements.  Matches CDNA5 A/B fragment K-striping when the caller
// passes  base + (lane>>4)*8.
static __device__ inline v16bf frag16(const bf16* p) {
  V16U u;
  u.h[0] = *(const v8bf*)(p);
  u.h[1] = *(const v8bf*)(p + 16);
  return u.v;
}

static __device__ inline v8f wmma_bf16(v16bf a, v16bf b, v8f c) {
  return __builtin_amdgcn_wmma_f32_16x16x32_bf16(
      /*neg_a=*/false, a, /*neg_b=*/false, b,
      /*c_mod=*/(short)0, c, /*reuse_a=*/false, /*reuse_b=*/false);
}

// ---------------------------------------------------------------------------
// One-time: Wt[n*K + k] = bf16(W[k*N + n]).  64x64 tiles via LDS transpose.
// ---------------------------------------------------------------------------
__global__ __launch_bounds__(256)
void convert_transpose_w(const float* __restrict__ W, bf16* __restrict__ Wt,
                         int K, int N)
{
  __shared__ bf16 T[64][72];        // [n][k], padded
  const int tid = threadIdx.x;
  const int n0  = blockIdx.x * 64;
  const int k0  = blockIdx.y * 64;

  #pragma unroll
  for (int v = 0; v < 4; ++v) {
    const int c    = tid + v * 256;       // 0..1023
    const int krow = c >> 4;              // 0..63
    const int col4 = c & 15;              // 0..15
    const float4 w = *(const float4*)&W[(long)(k0 + krow) * N + n0 + col4 * 4];
    T[col4 * 4 + 0][krow] = (bf16)w.x;
    T[col4 * 4 + 1][krow] = (bf16)w.y;
    T[col4 * 4 + 2][krow] = (bf16)w.z;
    T[col4 * 4 + 3][krow] = (bf16)w.w;
  }
  __syncthreads();
  #pragma unroll
  for (int v = 0; v < 2; ++v) {
    const int c    = tid + v * 256;       // 0..511
    const int nrow = c >> 3;              // 0..63
    const int part = c & 7;               // 0..7
    *(v8bf*)&Wt[(long)(n0 + nrow) * K + k0 + part * 8] =
        *(const v8bf*)&T[nrow][part * 8];
  }
}

// ---------------------------------------------------------------------------
// GEMM: C = A(MxK) * W(KxN) + bias, with W given pre-transposed as
// Bt (NxK bf16).  A is fp32 (Af) or bf16 (Ab), row-major.
// mode: 0 = bf16 C[m*N+n]; 1 = bf16 transposed Ct[n*M+m] (V); 2 = fp32.
// 256 threads = 8 waves; tile 64(M) x 128(N), K-step 32; double-buffered LDS,
// next tile's global loads issued before the current tile's WMMAs, and all
// fragment LDS loads issued as one clause so dscnt waits can stagger.
// ---------------------------------------------------------------------------
__global__ __launch_bounds__(256)
void gemm_bf16_wmma(const float* __restrict__ Af, const bf16* __restrict__ Ab,
                    const bf16* __restrict__ Bt, const float* __restrict__ bias,
                    bf16* __restrict__ Cb, float* __restrict__ Cf,
                    int M, int N, int K, int mode)
{
  __shared__ bf16 As[2][64][32];     // A tile, row-major (K contiguous)
  __shared__ bf16 Bs[2][128][40];    // B tile rows = n, padded (bank-safe)

  const int tid  = threadIdx.x;
  const int bn   = blockIdx.x;
  const int bm   = blockIdx.y;
  const int wid  = tid >> 5;
  const int lane = tid & 31;
  const int wm   = wid & 3;          // 4 waves along M (16 rows each)
  const int wn   = wid >> 2;         // 2 waves along N (64 cols each)
  const int lm   = lane & 15;
  const int lh   = lane >> 4;

  v8f acc[4] = { {0,0,0,0,0,0,0,0}, {0,0,0,0,0,0,0,0},
                 {0,0,0,0,0,0,0,0}, {0,0,0,0,0,0,0,0} };

  const int  arow  = tid >> 2;             // 0..63
  const int  acol  = (tid & 3) << 3;       // 0,8,16,24
  const long gArow = (long)(bm * 64 + arow);

  // ---- prologue: stage K-tile 0 into buffer 0 ----
  if (Af) {
    const float4 f0 = *(const float4*)&Af[gArow * K + acol];
    const float4 f1 = *(const float4*)&Af[gArow * K + acol + 4];
    v8bf pk;
    pk[0]=(bf16)f0.x; pk[1]=(bf16)f0.y; pk[2]=(bf16)f0.z; pk[3]=(bf16)f0.w;
    pk[4]=(bf16)f1.x; pk[5]=(bf16)f1.y; pk[6]=(bf16)f1.z; pk[7]=(bf16)f1.w;
    *(v8bf*)&As[0][arow][acol] = pk;
  } else {
    *(v8bf*)&As[0][arow][acol] = *(const v8bf*)&Ab[gArow * K + acol];
  }
  #pragma unroll
  for (int v = 0; v < 2; ++v) {
    const int c    = tid + v * 256;        // 0..511
    const int row  = c >> 2;               // 0..127
    const int part = c & 3;                // 0..3
    *(v8bf*)&Bs[0][row][part * 8] =
        *(const v8bf*)&Bt[(long)(bn * 128 + row) * K + part * 8];
  }
  __syncthreads();

  int cur = 0;
  for (int k0 = 0; k0 < K; k0 += 32) {
    const int  nxt = cur ^ 1;
    const bool hn  = (k0 + 32) < K;

    // ---- issue next tile's global loads (hide behind WMMAs) ----
    float4 fa0, fa1; v8bf ra; v8bf rb[2];
    if (hn) {
      const int kn = k0 + 32;
      if (Af) {
        fa0 = *(const float4*)&Af[gArow * K + kn + acol];
        fa1 = *(const float4*)&Af[gArow * K + kn + acol + 4];
      } else {
        ra = *(const v8bf*)&Ab[gArow * K + kn + acol];
      }
      #pragma unroll
      for (int v = 0; v < 2; ++v) {
        const int c    = tid + v * 256;
        const int row  = c >> 2;
        const int part = c & 3;
        rb[v] = *(const v8bf*)&Bt[(long)(bn * 128 + row) * K + kn + part * 8];
      }
      if (kn + 32 < K) {   // prefetch K+2 tile (global_prefetch_b8)
        __builtin_prefetch(&Bt[(long)(bn * 128 + (tid >> 2)) * K + kn + 32], 0, 1);
        if (Af) __builtin_prefetch(&Af[gArow * K + kn + 32], 0, 1);
        else    __builtin_prefetch(&Ab[gArow * K + kn + 32], 0, 1);
      }
    }

    // ---- compute on current buffer: issue all fragment loads first so the
    //      compiler staggers s_wait_dscnt, then run 4 WMMAs back-to-back ----
    const v16bf a = frag16(&As[cur][wm * 16 + lm][lh * 8]);
    v16bf bf4[4];
    #pragma unroll
    for (int j = 0; j < 4; ++j)
      bf4[j] = frag16(&Bs[cur][wn * 64 + j * 16 + lm][lh * 8]);
    #pragma unroll
    for (int j = 0; j < 4; ++j)
      acc[j] = wmma_bf16(a, bf4[j], acc[j]);

    // ---- store staged registers into the other buffer ----
    if (hn) {
      if (Af) {
        v8bf pk;
        pk[0]=(bf16)fa0.x; pk[1]=(bf16)fa0.y; pk[2]=(bf16)fa0.z; pk[3]=(bf16)fa0.w;
        pk[4]=(bf16)fa1.x; pk[5]=(bf16)fa1.y; pk[6]=(bf16)fa1.z; pk[7]=(bf16)fa1.w;
        *(v8bf*)&As[nxt][arow][acol] = pk;
      } else {
        *(v8bf*)&As[nxt][arow][acol] = ra;
      }
      #pragma unroll
      for (int v = 0; v < 2; ++v) {
        const int c    = tid + v * 256;
        const int row  = c >> 2;
        const int part = c & 3;
        *(v8bf*)&Bs[nxt][row][part * 8] = rb[v];
      }
    }
    __syncthreads();
    cur = nxt;
  }

  // ---- epilogue: bias + store ----
  #pragma unroll
  for (int j = 0; j < 4; ++j) {
    const int   n  = bn * 128 + wn * 64 + j * 16 + lm;
    const float bv = bias[n];
    #pragma unroll
    for (int r = 0; r < 8; ++r) {
      const int   m   = bm * 64 + wm * 16 + r + 8 * lh;
      const float val = acc[j][r] + bv;
      if (mode == 0)      Cb[(long)m * N + n] = (bf16)val;
      else if (mode == 1) Cb[(long)n * M + m] = (bf16)val;   // transposed (V)
      else                Cf[(long)m * N + n] = val;
    }
  }
}

// ---------------------------------------------------------------------------
// Flash attention (causal).  Score = (Qr Kr^T + Qi Ki^T)/8, softmax, O = P V
// for both Vr and Vi.  128 threads = 4 waves; each wave owns 16 query rows;
// key blocks of 32 (= one WMMA K step for P*V).  VrT/ViT are (D_MODEL x BT).
// Or/Oi may alias Qr/Qi: each wave reads only its own Q rows (into registers)
// before writing the same rows at the end.
// ---------------------------------------------------------------------------
__global__ __launch_bounds__(128)
void attn_flash_wmma(const bf16* Qr, const bf16* Qi,
                     const bf16* Kr, const bf16* Ki,
                     const bf16* VrT, const bf16* ViT,
                     bf16* Or, bf16* Oi)
{
  __shared__ bf16 Plds[4][16][40];   // per-wave 16x32 P tile (bf16), padded

  const int tid  = threadIdx.x;
  const int wid  = tid >> 5;
  const int lane = tid & 31;
  const int lm   = lane & 15;
  const int lh   = lane >> 4;

  const int bh = blockIdx.y;
  const int b  = bh / NUM_HEADS;
  const int h  = bh % NUM_HEADS;
  const int qb = blockIdx.x * 64 + wid * 16;     // query tile base in sequence
  const int hc = h * HEAD_DIM;
  const long qrow = (long)(b * SEQ + qb + lm);   // A-fragment row for this lane

  // Q fragments pinned in registers for the whole key loop (2 K-steps of 32)
  v16bf aqr[2], aqi[2];
  #pragma unroll
  for (int ks = 0; ks < 2; ++ks) {
    aqr[ks] = frag16(&Qr[qrow * D_MODEL + hc + ks * 32 + lh * 8]);
    aqi[ks] = frag16(&Qi[qrow * D_MODEL + hc + ks * 32 + lh * 8]);
  }

  v8f o_r[4] = { {0,0,0,0,0,0,0,0}, {0,0,0,0,0,0,0,0},
                 {0,0,0,0,0,0,0,0}, {0,0,0,0,0,0,0,0} };
  v8f o_i[4] = { {0,0,0,0,0,0,0,0}, {0,0,0,0,0,0,0,0},
                 {0,0,0,0,0,0,0,0}, {0,0,0,0,0,0,0,0} };
  float ms[8], ls[8];
  #pragma unroll
  for (int r = 0; r < 8; ++r) { ms[r] = -1e30f; ls[r] = 0.0f; }

  for (int s0 = 0; s0 <= qb + 15; s0 += 32) {
    // prefetch next key block (global_prefetch_b8)
    const int sn = s0 + 32;
    if (sn <= qb + 15) {
      __builtin_prefetch(&Kr[(long)(b * SEQ + sn + lm) * D_MODEL + hc], 0, 1);
      __builtin_prefetch(&Ki[(long)(b * SEQ + sn + lm) * D_MODEL + hc], 0, 1);
      __builtin_prefetch(&VrT[(long)(hc + lm) * BT + b * SEQ + sn], 0, 1);
      __builtin_prefetch(&ViT[(long)(hc + lm) * BT + b * SEQ + sn], 0, 1);
    }

    // ---- S tile 16x32: two 16x16 accumulators, 4 WMMAs each ----
    v8f sa[2] = { {0,0,0,0,0,0,0,0}, {0,0,0,0,0,0,0,0} };
    #pragma unroll
    for (int kh = 0; kh < 2; ++kh) {
      const long krow = (long)(b * SEQ + s0 + kh * 16 + lm);
      #pragma unroll
      for (int ks = 0; ks < 2; ++ks) {
        const v16bf bkr = frag16(&Kr[krow * D_MODEL + hc + ks * 32 + lh * 8]);
        sa[kh] = wmma_bf16(aqr[ks], bkr, sa[kh]);
        const v16bf bki = frag16(&Ki[krow * D_MODEL + hc + ks * 32 + lh * 8]);
        sa[kh] = wmma_bf16(aqi[ks], bki, sa[kh]);
      }
    }

    // ---- online softmax (row == C/D register index + 8*lh; 16 lanes/row) ----
    #pragma unroll
    for (int r = 0; r < 8; ++r) {
      const int q = qb + r + 8 * lh;
      float s0v = sa[0][r] * 0.125f;            // 1/sqrt(64)
      float s1v = sa[1][r] * 0.125f;
      if (s0 + lm > q)      s0v = -1e30f;       // causal mask
      if (s0 + 16 + lm > q) s1v = -1e30f;
      float mr = fmaxf(s0v, s1v);
      #pragma unroll
      for (int off = 1; off < 16; off <<= 1)
        mr = fmaxf(mr, __shfl_xor(mr, off, 16));
      const float mnew = fmaxf(ms[r], mr);
      const float p0   = __expf(s0v - mnew);
      const float p1   = __expf(s1v - mnew);
      float rs = p0 + p1;
      #pragma unroll
      for (int off = 1; off < 16; off <<= 1)
        rs += __shfl_xor(rs, off, 16);
      const float corr = __expf(ms[r] - mnew);
      ls[r] = ls[r] * corr + rs;
      ms[r] = mnew;
      #pragma unroll
      for (int j = 0; j < 4; ++j) { o_r[j][r] *= corr; o_i[j][r] *= corr; }
      // C-layout -> LDS (so we can re-read as an A fragment)
      Plds[wid][r + 8 * lh][lm]      = (bf16)p0;
      Plds[wid][r + 8 * lh][16 + lm] = (bf16)p1;
    }

    // ---- P (16x32) as A fragment; O += P * V ----
    const v16bf ap = frag16(&Plds[wid][lm][lh * 8]);
    #pragma unroll
    for (int j = 0; j < 4; ++j) {
      const long vrow = (long)(hc + j * 16 + lm) * BT + (long)(b * SEQ + s0);
      const v16bf bvr = frag16(&VrT[vrow + lh * 8]);
      o_r[j] = wmma_bf16(ap, bvr, o_r[j]);
      const v16bf bvi = frag16(&ViT[vrow + lh * 8]);
      o_i[j] = wmma_bf16(ap, bvi, o_i[j]);
    }
  }

  // ---- normalize and store (overwrites this wave's own Q rows) ----
  float inv[8];
  #pragma unroll
  for (int r = 0; r < 8; ++r) inv[r] = 1.0f / ls[r];
  #pragma unroll
  for (int j = 0; j < 4; ++j) {
    #pragma unroll
    for (int r = 0; r < 8; ++r) {
      const long row = (long)(b * SEQ + qb + r + 8 * lh);
      Or[row * D_MODEL + hc + j * 16 + lm] = (bf16)(o_r[j][r] * inv[r]);
      Oi[row * D_MODEL + hc + j * 16 + lm] = (bf16)(o_i[j][r] * inv[r]);
    }
  }
}

// ---------------------------------------------------------------------------
extern "C" void kernel_launch(void* const* d_in, const int* in_sizes, int n_in,
                              void* d_out, int out_size, void* d_ws, size_t ws_size,
                              hipStream_t stream)
{
  (void)in_sizes; (void)n_in; (void)out_size; (void)ws_size;

  const float* x_real = (const float*)d_in[0];
  const float* x_imag = (const float*)d_in[1];
  const float* Wqr = (const float*)d_in[2];  const float* bqr = (const float*)d_in[3];
  const float* Wqi = (const float*)d_in[4];  const float* bqi = (const float*)d_in[5];
  const float* Wkr = (const float*)d_in[6];  const float* bkr = (const float*)d_in[7];
  const float* Wki = (const float*)d_in[8];  const float* bki = (const float*)d_in[9];
  const float* Wvr = (const float*)d_in[10]; const float* bvr = (const float*)d_in[11];
  const float* Wvi = (const float*)d_in[12]; const float* bvi = (const float*)d_in[13];
  const float* Wor = (const float*)d_in[14]; const float* bor = (const float*)d_in[15];
  const float* Woi = (const float*)d_in[16]; const float* boi = (const float*)d_in[17];

  float* out_r = (float*)d_out;
  float* out_i = out_r + (size_t)BT * D_MODEL;

  // Workspace: 6 activation buffers (BT x D_MODEL bf16, 8 MB each) +
  // 8 transposed bf16 weight buffers (D_MODEL x D_MODEL, 2 MB each) = 64 MB.
  bf16* ws = (bf16*)d_ws;
  const size_t SZ  = (size_t)BT * D_MODEL;
  const size_t WSZ = (size_t)D_MODEL * D_MODEL;
  bf16* qr  = ws;           bf16* qi  = ws + 1 * SZ;
  bf16* kr  = ws + 2 * SZ;  bf16* ki  = ws + 3 * SZ;
  bf16* vrT = ws + 4 * SZ;  bf16* viT = ws + 5 * SZ;
  bf16* wt  = ws + 6 * SZ;
  bf16* wtQr = wt + 0 * WSZ; bf16* wtQi = wt + 1 * WSZ;
  bf16* wtKr = wt + 2 * WSZ; bf16* wtKi = wt + 3 * WSZ;
  bf16* wtVr = wt + 4 * WSZ; bf16* wtVi = wt + 5 * WSZ;
  bf16* wtOr = wt + 6 * WSZ; bf16* wtOi = wt + 7 * WSZ;

  // One-time weight convert/transpose (fp32 KxN -> bf16 NxK).
  dim3 tGrid(D_MODEL / 64, D_MODEL / 64), tBlk(256);
  convert_transpose_w<<<tGrid, tBlk, 0, stream>>>(Wqr, wtQr, D_MODEL, D_MODEL);
  convert_transpose_w<<<tGrid, tBlk, 0, stream>>>(Wqi, wtQi, D_MODEL, D_MODEL);
  convert_transpose_w<<<tGrid, tBlk, 0, stream>>>(Wkr, wtKr, D_MODEL, D_MODEL);
  convert_transpose_w<<<tGrid, tBlk, 0, stream>>>(Wki, wtKi, D_MODEL, D_MODEL);
  convert_transpose_w<<<tGrid, tBlk, 0, stream>>>(Wvr, wtVr, D_MODEL, D_MODEL);
  convert_transpose_w<<<tGrid, tBlk, 0, stream>>>(Wvi, wtVi, D_MODEL, D_MODEL);
  convert_transpose_w<<<tGrid, tBlk, 0, stream>>>(Wor, wtOr, D_MODEL, D_MODEL);
  convert_transpose_w<<<tGrid, tBlk, 0, stream>>>(Woi, wtOi, D_MODEL, D_MODEL);

  dim3 gGrid(D_MODEL / 128, BT / 64), gBlk(256);

  // Projections (fp32 inputs -> bf16 activations); V stored transposed.
  gemm_bf16_wmma<<<gGrid, gBlk, 0, stream>>>(x_real, nullptr, wtQr, bqr, qr,  nullptr, BT, D_MODEL, D_MODEL, 0);
  gemm_bf16_wmma<<<gGrid, gBlk, 0, stream>>>(x_imag, nullptr, wtQi, bqi, qi,  nullptr, BT, D_MODEL, D_MODEL, 0);
  gemm_bf16_wmma<<<gGrid, gBlk, 0, stream>>>(x_real, nullptr, wtKr, bkr, kr,  nullptr, BT, D_MODEL, D_MODEL, 0);
  gemm_bf16_wmma<<<gGrid, gBlk, 0, stream>>>(x_imag, nullptr, wtKi, bki, ki,  nullptr, BT, D_MODEL, D_MODEL, 0);
  gemm_bf16_wmma<<<gGrid, gBlk, 0, stream>>>(x_real, nullptr, wtVr, bvr, vrT, nullptr, BT, D_MODEL, D_MODEL, 1);
  gemm_bf16_wmma<<<gGrid, gBlk, 0, stream>>>(x_imag, nullptr, wtVi, bvi, viT, nullptr, BT, D_MODEL, D_MODEL, 1);

  // Flash attention; outputs reuse qr/qi buffers (row-safe, see kernel).
  dim3 aGrid(SEQ / 64, BATCH * NUM_HEADS), aBlk(128);
  attn_flash_wmma<<<aGrid, aBlk, 0, stream>>>(qr, qi, kr, ki, vrT, viT, qr, qi);

  // Output projections -> fp32 d_out (final_r then final_i, concatenated).
  gemm_bf16_wmma<<<gGrid, gBlk, 0, stream>>>(nullptr, qr, wtOr, bor, nullptr, out_r, BT, D_MODEL, D_MODEL, 2);
  gemm_bf16_wmma<<<gGrid, gBlk, 0, stream>>>(nullptr, qi, wtOi, boi, nullptr, out_i, BT, D_MODEL, D_MODEL, 2);
}
